// DeeperGCN_71837622992996
// MI455X (gfx1250) — matrix operand order repaired and checked by reading, not compiled, not run on hardware
//
#include <hip/hip_runtime.h>
#include <math.h>

// ---------------- problem constants ----------------
#define NN      50000      // nodes
#define NE      600000     // edges
#define NG      32         // graphs
#define IN_F    16
#define HID     128
#define HID2    256        // HID * EXP
#define OUT_F   10
#define EDGE_DIM 16
#define BN_EPSF 1e-5f
#define GEN_EPSF 1e-7f
#define NB_STAT 200        // blocks for deterministic BN reduction

typedef __bf16 bf16;
typedef __attribute__((ext_vector_type(16))) __bf16 v16bf;
typedef __attribute__((ext_vector_type(4)))  __bf16 bf16x4;
typedef __attribute__((ext_vector_type(8)))  float  v8f;
typedef int v4i __attribute__((vector_size(16)));

// ---------------- gfx1250 async load-to-LDS (guarded) ----------------
#ifndef __has_builtin
#define __has_builtin(x) 0
#endif
#if __has_builtin(__builtin_amdgcn_global_load_async_to_lds_b128)
#define USE_ASYNC_LDS 1
#else
#define USE_ASYNC_LDS 0
#endif

__device__ __forceinline__ void cp16_to_lds(const float* __restrict__ g, float* l) {
#if USE_ASYNC_LDS
  __builtin_amdgcn_global_load_async_to_lds_b128(
      (__attribute__((address_space(1))) v4i*)(void*)g,
      (__attribute__((address_space(3))) v4i*)(void*)l, 0, 0);
#else
  *(float4*)l = *(const float4*)g;
#endif
}

__device__ __forceinline__ void wait_async_lds() {
#if USE_ASYNC_LDS
#if __has_builtin(__builtin_amdgcn_s_wait_asynccnt)
  __builtin_amdgcn_s_wait_asynccnt(0);
#else
  asm volatile("s_wait_asynccnt 0" ::: "memory");
#endif
#endif
}

// ---------------- WMMA fragment loaders from global (bf16, wave32) ----------
// A: 16x32 (MxK), ISA 7.12.2 layout. f32->bf16 converted in regs.
__device__ __forceinline__ v16bf load_a_frag(const float* __restrict__ A, int lda,
                                             int m0, int kk, int Kmax) {
  const int lane = threadIdx.x & 31;
  const int m    = m0 + (lane & 15);
  const int half = lane >> 4;
  v16bf a;
#pragma unroll
  for (int v = 0; v < 8; ++v) {
    const int k0 = kk + ((v < 4) ? (2 * v) : (16 + 2 * (v - 4))) + 8 * half;
    float fx = 0.0f, fy = 0.0f;
    if (k0 < Kmax) {
      const float2 f = *(const float2*)(A + (size_t)m * lda + k0);
      fx = f.x; fy = f.y;
    }
    a[2 * v]     = (bf16)fx;
    a[2 * v + 1] = (bf16)fy;
  }
  return a;
}

// B: 32x16 (KxN): lane = N column; same K order as A (lanes 0-15 K=0..7,16..23; lanes 16-31 K=8..15,24..31).
__device__ __forceinline__ v16bf load_b_frag(const float* __restrict__ B, int ldb,
                                             int n0, int kk, int Kmax) {
  const int lane = threadIdx.x & 31;
  const int n    = n0 + (lane & 15);
  const int half = lane >> 4;
  v16bf b;
#pragma unroll
  for (int v = 0; v < 8; ++v) {
    const int k = kk + ((v < 4) ? (2 * v) : (16 + 2 * (v - 4))) + 8 * half;
    float fx = 0.0f, fy = 0.0f;
    if (k < Kmax) {
      fx = B[(size_t)k * ldb + n];
      fy = B[(size_t)(k + 1) * ldb + n];
    }
    b[2 * v]     = (bf16)fx;
    b[2 * v + 1] = (bf16)fy;
  }
  return b;
}

// ---------------- small-K GEMMs (K=16 zero-padded to 32) ----------------
// h = x @ W_in + b   (M=NN, N=128). grid(3125,2), blk 128 (4 waves = 4 n-tiles).
__global__ __launch_bounds__(128) void fc_in_kernel(const float* __restrict__ X,
    const float* __restrict__ W, const float* __restrict__ bias, float* __restrict__ H) {
  const int wave = threadIdx.x >> 5;
  const int m0 = blockIdx.x * 16;
  const int n0 = (blockIdx.y * 4 + wave) * 16;
  v8f acc = {};
  v16bf a = load_a_frag(X, IN_F, m0, 0, IN_F);
  v16bf b = load_b_frag(W, HID, n0, 0, IN_F);
  acc = __builtin_amdgcn_wmma_f32_16x16x32_bf16(false, a, false, b, (short)0, acc, false, false);
  const int lane = threadIdx.x & 31;
  const int nc = n0 + (lane & 15), half = lane >> 4;
#pragma unroll
  for (int v = 0; v < 8; ++v)
    H[(size_t)(m0 + v + 8 * half) * HID + nc] = acc[v] + bias[nc];
}

// e_bf16 = edge_attr @ We + be  (M=NE, N=128). grid(37500,2), blk 128.
__global__ __launch_bounds__(128) void edge_gemm_kernel(const float* __restrict__ EA,
    const float* __restrict__ We, const float* __restrict__ be, bf16* __restrict__ Ebf) {
  const int wave = threadIdx.x >> 5;
  const int m0 = blockIdx.x * 16;
  const int n0 = (blockIdx.y * 4 + wave) * 16;
  v8f acc = {};
  v16bf a = load_a_frag(EA, EDGE_DIM, m0, 0, EDGE_DIM);
  v16bf b = load_b_frag(We, HID, n0, 0, EDGE_DIM);
  acc = __builtin_amdgcn_wmma_f32_16x16x32_bf16(false, a, false, b, (short)0, acc, false, false);
  const int lane = threadIdx.x & 31;
  const int nc = n0 + (lane & 15), half = lane >> 4;
#pragma unroll
  for (int v = 0; v < 8; ++v)
    Ebf[(size_t)(m0 + v + 8 * half) * HID + nc] = (bf16)(acc[v] + be[nc]);
}

// ---------------- LDS-staged GEMM for the MLP (K in {128,256}) ----------------
// Block = 128 thr (4 waves). Tile: 16 rows x 64 cols; wave w owns cols w*16..+15.
// A slab (16 x 128) and B slab (128 x 64) staged per K-chunk via async-to-LDS.
// MODE 0: C = acc + bias ; MODE 1: C += acc + bias.
template <int K, int NCOLS, int MODE>
__global__ __launch_bounds__(128) void gemm_lds_kernel(const float* __restrict__ A,
    const float* __restrict__ B, const float* __restrict__ bias, float* __restrict__ C) {
  constexpr int KC = 128;
  __shared__ float sA[16 * KC];   // 8 KB
  __shared__ float sB[KC * 64];   // 32 KB
  const int t = threadIdx.x;
  const int wave = t >> 5;
  const int lane = t & 31;
  const int m0 = blockIdx.x * 16;
  const int nblk = blockIdx.y * 64;
  const int mrow = lane & 15, half = lane >> 4;
  const int nloc = wave * 16 + (lane & 15);
  v8f acc = {};
#pragma unroll
  for (int kc0 = 0; kc0 < K; kc0 += KC) {
    if (kc0) __syncthreads();   // protect LDS reuse across chunks
    // stage A slab: 16 x KC floats = 512 float4, 4 per thread
#pragma unroll
    for (int i = 0; i < 4; ++i) {
      const int f = t + i * 128;
      const int r = f >> 5;            // f / (KC/4)
      const int c4 = (f & 31) * 4;     // (f % 32) * 4
      cp16_to_lds(A + (size_t)(m0 + r) * K + kc0 + c4, &sA[r * KC + c4]);
    }
    // stage B slab: KC x 64 floats = 2048 float4, 16 per thread
#pragma unroll
    for (int i = 0; i < 16; ++i) {
      const int f = t + i * 128;
      const int r = f >> 4;            // f / 16
      const int c4 = (f & 15) * 4;
      cp16_to_lds(B + (size_t)(kc0 + r) * NCOLS + nblk + c4, &sB[r * 64 + c4]);
    }
    wait_async_lds();
    __syncthreads();
#pragma unroll
    for (int kk = 0; kk < KC; kk += 32) {
      v16bf a, b;
#pragma unroll
      for (int v = 0; v < 8; ++v) {
        const int k0 = kk + ((v < 4) ? (2 * v) : (16 + 2 * (v - 4))) + 8 * half;
        const float2 f2 = *(const float2*)&sA[mrow * KC + k0];
        a[2 * v]     = (bf16)f2.x;
        a[2 * v + 1] = (bf16)f2.y;
        const int kb = kk + ((v < 4) ? (2 * v) : (16 + 2 * (v - 4))) + 8 * half;
        b[2 * v]     = (bf16)sB[kb * 64 + nloc];
        b[2 * v + 1] = (bf16)sB[(kb + 1) * 64 + nloc];
      }
      acc = __builtin_amdgcn_wmma_f32_16x16x32_bf16(false, a, false, b, (short)0, acc, false, false);
    }
  }
  const int nc = nblk + nloc;
#pragma unroll
  for (int v = 0; v < 8; ++v) {
    float* p = C + (size_t)(m0 + v + 8 * half) * NCOLS + nc;
    const float val = acc[v] + bias[nc];
    if (MODE) *p += val; else *p = val;
  }
}

// ---------------- BatchNorm (training stats, deterministic 2-stage) ----------
__global__ void bn_stats_kernel(const float* __restrict__ Z, int C, int Nrows,
                                float* __restrict__ psum, float* __restrict__ psq) {
  __shared__ float sh[256], sh2[256];
  const int t = threadIdx.x;
  const int RP = 256 / C;              // 2 for C=128, 1 for C=256
  const int c = t % C, ro = t / C;
  const int rowsPer = (Nrows + NB_STAT - 1) / NB_STAT;
  const int r0 = blockIdx.x * rowsPer;
  int r1 = r0 + rowsPer; if (r1 > Nrows) r1 = Nrows;
  float s = 0.0f, s2 = 0.0f;
  for (int r = r0 + ro; r < r1; r += RP) {
    const float v = Z[(size_t)r * C + c];
    s += v; s2 += v * v;
  }
  sh[t] = s; sh2[t] = s2;
  __syncthreads();
  if (t < C) {
    for (int i = 1; i < RP; ++i) { s += sh[t + i * C]; s2 += sh2[t + i * C]; }
    psum[blockIdx.x * C + t] = s;
    psq [blockIdx.x * C + t] = s2;
  }
}

__global__ void bn_finalize_kernel(const float* __restrict__ psum, const float* __restrict__ psq,
                                   int C, float invN, float* __restrict__ mean,
                                   float* __restrict__ rstd) {
  const int c = blockIdx.x * blockDim.x + threadIdx.x;
  if (c >= C) return;
  float s = 0.0f, s2 = 0.0f;
  for (int i = 0; i < NB_STAT; ++i) { s += psum[i * C + c]; s2 += psq[i * C + c]; }
  const float mu = s * invN;
  const float var = s2 * invN - mu * mu;     // biased, matches z.var(0)
  mean[c] = mu;
  rstd[c] = rsqrtf(var + BN_EPSF);
}

__global__ void bn_apply_relu_kernel(const float* __restrict__ Z, int cmask, size_t total,
                                     const float* __restrict__ mean, const float* __restrict__ rstd,
                                     const float* __restrict__ g, const float* __restrict__ b,
                                     float* __restrict__ Y) {
  const size_t i = (size_t)blockIdx.x * blockDim.x + threadIdx.x;
  if (i >= total) return;
  const int c = (int)(i & (size_t)cmask);
  const float v = (Z[i] - mean[c]) * rstd[c] * g[c] + b[c];
  Y[i] = fmaxf(v, 0.0f);
}

// ---------------- edge segment-softmax passes (4 channels / thread) ----------
// msg = relu(hn[src]+e)+eps > 0 and t=1>0, so logits > 0: u32 atomicMax over
// float bits is order-preserving; mx buffer initialized to 0x0.
__global__ void edge_pass1_max(const int* __restrict__ ei, const bf16* __restrict__ ebf,
                               const float* __restrict__ hn, const float* __restrict__ tptr,
                               unsigned int* __restrict__ mxbits) {
  const size_t idx = (size_t)blockIdx.x * blockDim.x + threadIdx.x;
  if (idx >= (size_t)NE * 32) return;
  const int e = (int)(idx >> 5);
  const int q = ((int)idx & 31) * 4;
  const int s = ei[e], d = ei[NE + e];
  const float tt = tptr[0];
  const bf16x4 ev = *(const bf16x4*)(ebf + (size_t)e * HID + q);
  const float4 hv = *(const float4*)(hn + (size_t)s * HID + q);
  unsigned int* mp = mxbits + (size_t)d * HID + q;
  atomicMax(mp + 0, __float_as_uint(tt * (fmaxf(hv.x + (float)ev[0], 0.0f) + GEN_EPSF)));
  atomicMax(mp + 1, __float_as_uint(tt * (fmaxf(hv.y + (float)ev[1], 0.0f) + GEN_EPSF)));
  atomicMax(mp + 2, __float_as_uint(tt * (fmaxf(hv.z + (float)ev[2], 0.0f) + GEN_EPSF)));
  atomicMax(mp + 3, __float_as_uint(tt * (fmaxf(hv.w + (float)ev[3], 0.0f) + GEN_EPSF)));
}

__global__ void edge_pass2_den(const int* __restrict__ ei, const bf16* __restrict__ ebf,
                               const float* __restrict__ hn, const float* __restrict__ tptr,
                               const unsigned int* __restrict__ mxbits, float* __restrict__ den) {
  const size_t idx = (size_t)blockIdx.x * blockDim.x + threadIdx.x;
  if (idx >= (size_t)NE * 32) return;
  const int e = (int)(idx >> 5);
  const int q = ((int)idx & 31) * 4;
  const int s = ei[e], d = ei[NE + e];
  const float tt = tptr[0];
  const bf16x4 ev = *(const bf16x4*)(ebf + (size_t)e * HID + q);
  const float4 hv = *(const float4*)(hn + (size_t)s * HID + q);
  const size_t o = (size_t)d * HID + q;
  const float4 mx = *(const float4*)((const float*)mxbits + o);
  float* dp = den + o;
  atomicAdd(dp + 0, expf(tt * (fmaxf(hv.x + (float)ev[0], 0.0f) + GEN_EPSF) - mx.x));
  atomicAdd(dp + 1, expf(tt * (fmaxf(hv.y + (float)ev[1], 0.0f) + GEN_EPSF) - mx.y));
  atomicAdd(dp + 2, expf(tt * (fmaxf(hv.z + (float)ev[2], 0.0f) + GEN_EPSF) - mx.z));
  atomicAdd(dp + 3, expf(tt * (fmaxf(hv.w + (float)ev[3], 0.0f) + GEN_EPSF) - mx.w));
}

__global__ void edge_pass3_agg(const int* __restrict__ ei, const bf16* __restrict__ ebf,
                               const float* __restrict__ hn, const float* __restrict__ tptr,
                               const unsigned int* __restrict__ mxbits,
                               const float* __restrict__ den, float* __restrict__ aggr) {
  const size_t idx = (size_t)blockIdx.x * blockDim.x + threadIdx.x;
  if (idx >= (size_t)NE * 32) return;
  const int e = (int)(idx >> 5);
  const int q = ((int)idx & 31) * 4;
  const int s = ei[e], d = ei[NE + e];
  const float tt = tptr[0];
  const bf16x4 ev = *(const bf16x4*)(ebf + (size_t)e * HID + q);
  const float4 hv = *(const float4*)(hn + (size_t)s * HID + q);
  const size_t o = (size_t)d * HID + q;
  const float4 mx = *(const float4*)((const float*)mxbits + o);
  const float4 dn = *(const float4*)(den + o);
  float* ap = aggr + o;
  {
    const float m = fmaxf(hv.x + (float)ev[0], 0.0f) + GEN_EPSF;
    atomicAdd(ap + 0, (expf(tt * m - mx.x) / (dn.x + 1e-16f)) * m);
  }
  {
    const float m = fmaxf(hv.y + (float)ev[1], 0.0f) + GEN_EPSF;
    atomicAdd(ap + 1, (expf(tt * m - mx.y) / (dn.y + 1e-16f)) * m);
  }
  {
    const float m = fmaxf(hv.z + (float)ev[2], 0.0f) + GEN_EPSF;
    atomicAdd(ap + 2, (expf(tt * m - mx.z) / (dn.z + 1e-16f)) * m);
  }
  {
    const float m = fmaxf(hv.w + (float)ev[3], 0.0f) + GEN_EPSF;
    atomicAdd(ap + 3, (expf(tt * m - mx.w) / (dn.w + 1e-16f)) * m);
  }
}

// ---------------- misc elementwise ----------------
__global__ void fill_f32_kernel(float* __restrict__ p, float v, size_t n) {
  const size_t i = (size_t)blockIdx.x * blockDim.x + threadIdx.x;
  if (i < n) p[i] = v;
}

__global__ void add_kernel(float* __restrict__ dst, const float* __restrict__ src, size_t n) {
  const size_t i = (size_t)blockIdx.x * blockDim.x + threadIdx.x;
  if (i < n) dst[i] += src[i];
}

// ---------------- pooling + head ----------------
__global__ void pool_count_kernel(const int* __restrict__ batch, float* __restrict__ counts) {
  const int i = blockIdx.x * blockDim.x + threadIdx.x;
  if (i < NN) atomicAdd(&counts[batch[i]], 1.0f);
}

__global__ void pool_sum_kernel(const float* __restrict__ H, const int* __restrict__ batch,
                                float* __restrict__ pooled) {
  const size_t idx = (size_t)blockIdx.x * blockDim.x + threadIdx.x;
  if (idx >= (size_t)NN * HID) return;
  const int n = (int)(idx >> 7), c = (int)(idx & 127);
  atomicAdd(&pooled[batch[n] * HID + c], H[idx]);
}

__global__ void fc_out_kernel(const float* __restrict__ pooled, const float* __restrict__ counts,
                              const float* __restrict__ W, const float* __restrict__ b,
                              float* __restrict__ out) {
  __shared__ float sh[HID];
  const int g = blockIdx.x, t = threadIdx.x;
  const float cnt = fmaxf(counts[g], 1.0f);
  for (int c = t; c < HID; c += blockDim.x)
    sh[c] = fmaxf(pooled[g * HID + c] / cnt, 0.0f);
  __syncthreads();
  if (t < OUT_F) {
    float acc = b[t];
    for (int c = 0; c < HID; ++c) acc += sh[c] * W[c * OUT_F + t];
    out[g * OUT_F + t] = acc;
  }
}

// ---------------- host launch ----------------
extern "C" void kernel_launch(void* const* d_in, const int* in_sizes, int n_in,
                              void* d_out, int out_size, void* d_ws, size_t ws_size,
                              hipStream_t stream) {
  (void)in_sizes; (void)n_in; (void)out_size; (void)ws_size;
  // setup_inputs() insertion-order flattening
  const float* x      = (const float*)d_in[0];
  const int*   ei     = (const int*)  d_in[1];   // [2, NE]
  const float* ea     = (const float*)d_in[2];   // [NE, 16]
  const int*   batch  = (const int*)  d_in[3];
  const float* fcinW  = (const float*)d_in[4];
  const float* fcinb  = (const float*)d_in[5];
  const float* fcoutW = (const float*)d_in[6];
  const float* fcoutb = (const float*)d_in[7];
  // layers: 11 leaves each, starting at 8

  char* wp = (char*)d_ws;
  auto alloc = [&](size_t bytes) -> char* {
    char* p = wp; wp += (bytes + 255) & ~(size_t)255; return p;
  };
  const size_t NH  = (size_t)NN * HID;
  const size_t NH2 = (size_t)NN * HID2;
  const size_t EH  = (size_t)NE * HID;

  float* h    = (float*)alloc(NH * 4);
  float* hn   = (float*)alloc(NH * 4);
  float* mx   = (float*)alloc(3 * NH * 4);       // mx | den | aggr contiguous
  float* den  = mx + NH;
  float* aggr = den + NH;
  float* z1   = (float*)alloc(NH2 * 4);
  bf16*  ebf  = (bf16*) alloc(EH * 2);
  float* psum = (float*)alloc((size_t)NB_STAT * 256 * 4);
  float* psq  = (float*)alloc((size_t)NB_STAT * 256 * 4);
  float* mean = (float*)alloc(256 * 4);
  float* rstd = (float*)alloc(256 * 4);
  float* counts = (float*)alloc((NG + NG * HID) * 4);  // counts | pooled
  float* pooled = counts + NG;

  const dim3 blk128(128);
  const dim3 blk256(256);
  const int EBLK4 = (int)(((size_t)NE * 32 + 255) / 256);
  const int NBLK  = (int)((NH + 255) / 256);
  const int NBLK2 = (int)((NH2 + 255) / 256);

  // fc_in: h = x @ W + b   [WMMA]
  fc_in_kernel<<<dim3(NN / 16, 2), blk128, 0, stream>>>(x, fcinW, fcinb, h);

  for (int l = 0; l < 2; ++l) {
    const int base = 8 + 11 * l;
    const float* bng   = (const float*)d_in[base + 0];
    const float* bnb   = (const float*)d_in[base + 1];
    const float* We    = (const float*)d_in[base + 2];
    const float* be    = (const float*)d_in[base + 3];
    const float* tp    = (const float*)d_in[base + 4];
    const float* W1    = (const float*)d_in[base + 5];
    const float* b1    = (const float*)d_in[base + 6];
    const float* g1    = (const float*)d_in[base + 7];
    const float* beta1 = (const float*)d_in[base + 8];
    const float* W2    = (const float*)d_in[base + 9];
    const float* b2    = (const float*)d_in[base + 10];

    // hn = relu(BN(h))
    bn_stats_kernel<<<NB_STAT, blk256, 0, stream>>>(h, HID, NN, psum, psq);
    bn_finalize_kernel<<<1, blk256, 0, stream>>>(psum, psq, HID, 1.0f / NN, mean, rstd);
    bn_apply_relu_kernel<<<NBLK, blk256, 0, stream>>>(h, HID - 1, NH, mean, rstd, bng, bnb, hn);

    // e (bf16, L2-resident) = edge_attr @ We + be   [WMMA]
    edge_gemm_kernel<<<dim3(NE / 16, 2), blk128, 0, stream>>>(ea, We, be, ebf);

    // segment softmax aggregation (3 atomic passes, recompute msg each pass)
    fill_f32_kernel<<<(int)((3 * NH + 255) / 256), blk256, 0, stream>>>(mx, 0.0f, 3 * NH);
    edge_pass1_max<<<EBLK4, blk256, 0, stream>>>(ei, ebf, hn, tp, (unsigned int*)mx);
    edge_pass2_den<<<EBLK4, blk256, 0, stream>>>(ei, ebf, hn, tp, (const unsigned int*)mx, den);
    edge_pass3_agg<<<EBLK4, blk256, 0, stream>>>(ei, ebf, hn, tp, (const unsigned int*)mx, den, aggr);

    // out = aggr + hn  (in place in aggr)
    add_kernel<<<NBLK, blk256, 0, stream>>>(aggr, hn, NH);

    // MLP: z1 = out @ W1 + b1 ; z1 = relu(BN(z1)) ; h += z1 @ W2 + b2  [WMMA+LDS]
    gemm_lds_kernel<HID, HID2, 0><<<dim3(NN / 16, HID2 / 64), blk128, 0, stream>>>(aggr, W1, b1, z1);
    bn_stats_kernel<<<NB_STAT, blk256, 0, stream>>>(z1, HID2, NN, psum, psq);
    bn_finalize_kernel<<<1, blk256, 0, stream>>>(psum, psq, HID2, 1.0f / NN, mean, rstd);
    bn_apply_relu_kernel<<<NBLK2, blk256, 0, stream>>>(z1, HID2 - 1, NH2, mean, rstd, g1, beta1, z1);
    gemm_lds_kernel<HID2, HID, 1><<<dim3(NN / 16, HID / 64), blk128, 0, stream>>>(z1, W2, b2, h);
  }

  // global mean pool + relu + fc_out
  fill_f32_kernel<<<(NG + NG * HID + 255) / 256, blk256, 0, stream>>>(counts, 0.0f, NG + NG * HID);
  pool_count_kernel<<<(NN + 255) / 256, blk256, 0, stream>>>(batch, counts);
  pool_sum_kernel<<<NBLK, blk256, 0, stream>>>(h, batch, pooled);
  fc_out_kernel<<<NG, blk128, 0, stream>>>(pooled, counts, fcoutW, fcoutb, (float*)d_out);
}
